// GCNLayerV2_57947698758370
// MI455X (gfx1250) — matrix operand order repaired and checked by reading, not compiled, not run on hardware
//
#include <hip/hip_runtime.h>

typedef __attribute__((ext_vector_type(16))) __bf16        v16bf;
typedef __attribute__((ext_vector_type(8)))  float         v8f;
typedef __attribute__((ext_vector_type(4)))  float         f32x4;

union BF16Frag { unsigned int w[8]; unsigned short u[16]; v16bf v; };

#define LDS_PITCH        136                     // ushorts per row (128 + 8 pad)
#define LDS_ROW_BYTES    (LDS_PITCH * 2)         // 272 B -> rows 4 banks apart
#define LDS_PLANE        (128 * LDS_PITCH)       // ushorts per plane
#define LDS_PLANE_BYTES  (128 * LDS_ROW_BYTES)   // 34816 B
#define LDS_TOTAL_BYTES  (4 * LDS_PLANE_BYTES)   // 139264 B (dynamic LDS)

__device__ __forceinline__ unsigned int rne_bits(float f) {
    unsigned int b = __builtin_bit_cast(unsigned int, f);
    return b + 0x7FFFu + ((b >> 16) & 1u);
}
__device__ __forceinline__ float hi_part(unsigned int rounded) {
    return __builtin_bit_cast(float, rounded & 0xFFFF0000u);
}

// ---------------------------------------------------------------------------
// Transpose both 128x128 W matrices into 4 contiguous bf16 planes
// [wnh | wnl | wsh | wsl], each [n][k] with K contiguous (128 ushorts/row).
// hi = bf16(w), lo = bf16(w - hi)  (bf16x3 decomposition source data).
// ---------------------------------------------------------------------------
__global__ __launch_bounds__(256) void prep_wT(
        const float* __restrict__ Wn, const float* __restrict__ Ws,
        unsigned short* __restrict__ wplanes) {
    int idx = blockIdx.x * blockDim.x + threadIdx.x;   // 0..16383
    if (idx >= 128 * 128) return;
    int k = idx >> 7;
    int n = idx & 127;
    int t = n * 128 + k;

    unsigned short* wnh = wplanes;
    unsigned short* wnl = wplanes + 1 * 128 * 128;
    unsigned short* wsh = wplanes + 2 * 128 * 128;
    unsigned short* wsl = wplanes + 3 * 128 * 128;

    float a = Wn[idx];
    unsigned int ra = rne_bits(a);
    float ar = a - hi_part(ra);
    wnh[t] = (unsigned short)(ra >> 16);
    wnl[t] = (unsigned short)(rne_bits(ar) >> 16);

    float b = Ws[idx];
    unsigned int rb = rne_bits(b);
    float br = b - hi_part(rb);
    wsh[t] = (unsigned short)(rb >> 16);
    wsl[t] = (unsigned short)(rne_bits(br) >> 16);
}

// ---------------------------------------------------------------------------
// Fused dual GEMM: y = x @ W_nbrs, out = x @ W_self + bias (bf16x3 WMMA).
// Per block: async-DMA the 4 W planes (128 KB) global->LDS once
// (global_load_async_to_lds_b128 / ASYNCcnt), then all 8 waves read B
// fragments from LDS (padded pitch -> conflict-free ds_load_b128).
// Per wave: 4 K-tiles x 8 N-tiles x 2 matrices x 3 = 192 WMMAs.
// ---------------------------------------------------------------------------
__global__ __launch_bounds__(256) void gemm_dual_bf16x3(
        const float* __restrict__ x,
        const unsigned short* __restrict__ wplanes,
        const float* __restrict__ bias,
        float* __restrict__ y, float* __restrict__ out, int n_rows) {
    extern __shared__ unsigned short smem[];   // 4 padded planes, 136 KB

    const int lane  = threadIdx.x & 31;
    const int wave  = threadIdx.x >> 5;                  // 0..7
    const int row0  = blockIdx.x * 128 + wave * 16;
    const int mlo   = lane & 15;
    const int khalf = lane >> 4;                         // 0 or 1

    // ---- Stage W planes into LDS with async DMA (no VGPR round-trip).
    // 8192 16-byte chunks total -> 32 per thread.
    {
        const unsigned lds0  = __builtin_amdgcn_groupstaticsize();
        const unsigned long long gbase = (unsigned long long)wplanes;
#pragma unroll
        for (int i = 0; i < 32; ++i) {
            int g     = threadIdx.x + 256 * i;
            int plane = g >> 11;                 // 0..3
            int r     = (g >> 4) & 127;          // row
            int c     = g & 15;                  // 16B chunk in row
            unsigned voff = (unsigned)(plane * 32768 + r * 256 + c * 16);
            unsigned ldsa = lds0 +
                (unsigned)(plane * LDS_PLANE_BYTES + r * LDS_ROW_BYTES + c * 16);
            asm volatile("global_load_async_to_lds_b128 %0, %1, %2"
                         :: "v"(ldsa), "v"(voff), "s"(gbase) : "memory");
        }
        asm volatile("s_wait_asynccnt 0" ::: "memory");
    }
    __syncthreads();

    const unsigned short* snh = smem + 0 * LDS_PLANE;
    const unsigned short* snl = smem + 1 * LDS_PLANE;
    const unsigned short* ssh = smem + 2 * LDS_PLANE;
    const unsigned short* ssl = smem + 3 * LDS_PLANE;

    int arow = row0 + mlo;
    if (arow >= n_rows) arow = n_rows - 1;               // clamp (loads only)
    const float* xrow = x + (size_t)arow * 128;

    v8f zero = {0.f, 0.f, 0.f, 0.f, 0.f, 0.f, 0.f, 0.f};
    v8f accn[8], accs[8];
#pragma unroll
    for (int nt = 0; nt < 8; ++nt) { accn[nt] = zero; accs[nt] = zero; }

#pragma unroll
    for (int kt = 0; kt < 4; ++kt) {
        // ---- A fragment (contiguous K order, same as B): one contiguous 16-float
        // run per lane: K = kt*32 + khalf*16 + [0..15].
        const f32x4* p0 = (const f32x4*)(xrow + kt * 32 + khalf * 16);
        const f32x4* p1 = (const f32x4*)(xrow + kt * 32 + 8 + khalf * 16);
        f32x4 r0 = p0[0], r1 = p0[1], r2 = p1[0], r3 = p1[1];
        float buf[16] = {r0.x, r0.y, r0.z, r0.w, r1.x, r1.y, r1.z, r1.w,
                         r2.x, r2.y, r2.z, r2.w, r3.x, r3.y, r3.z, r3.w};
        BF16Frag a_hi, a_lo;
#pragma unroll
        for (int i = 0; i < 8; ++i) {
            float f0 = buf[2 * i], f1 = buf[2 * i + 1];
            unsigned int h0 = rne_bits(f0), h1 = rne_bits(f1);
            a_hi.w[i] = (h0 >> 16) | (h1 & 0xFFFF0000u);
            float l0 = f0 - hi_part(h0), l1 = f1 - hi_part(h1);
            unsigned int c0 = rne_bits(l0), c1 = rne_bits(l1);
            a_lo.w[i] = (c0 >> 16) | (c1 & 0xFFFF0000u);
        }

        const int koff = kt * 32 + khalf * 16;           // B K-group start
#pragma unroll
        for (int nt = 0; nt < 8; ++nt) {
            // ---- B fragment (16-bit B 32x16): N = nt*16 + mlo; 16
            // consecutive bf16 -> one 32B LDS vector load per plane.
            size_t roff = (size_t)(nt * 16 + mlo) * LDS_PITCH + koff;
            v16bf bnh = *(const v16bf*)(snh + roff);
            v16bf bnl = *(const v16bf*)(snl + roff);
            accn[nt] = __builtin_amdgcn_wmma_f32_16x16x32_bf16(
                false, a_hi.v, false, bnh, (short)0, accn[nt], false, false);
            accn[nt] = __builtin_amdgcn_wmma_f32_16x16x32_bf16(
                false, a_hi.v, false, bnl, (short)0, accn[nt], false, false);
            accn[nt] = __builtin_amdgcn_wmma_f32_16x16x32_bf16(
                false, a_lo.v, false, bnh, (short)0, accn[nt], false, false);

            v16bf bsh = *(const v16bf*)(ssh + roff);
            v16bf bsl = *(const v16bf*)(ssl + roff);
            accs[nt] = __builtin_amdgcn_wmma_f32_16x16x32_bf16(
                false, a_hi.v, false, bsh, (short)0, accs[nt], false, false);
            accs[nt] = __builtin_amdgcn_wmma_f32_16x16x32_bf16(
                false, a_hi.v, false, bsl, (short)0, accs[nt], false, false);
            accs[nt] = __builtin_amdgcn_wmma_f32_16x16x32_bf16(
                false, a_lo.v, false, bsh, (short)0, accs[nt], false, false);
        }
    }

    // ---- Epilogue: C/D 16x16 layout: VGPR r -> M = r + khalf*8, N = mlo.
#pragma unroll
    for (int r = 0; r < 8; ++r) {
        int row = row0 + r + khalf * 8;
        if (row < n_rows) {
            float* yrow = y   + (size_t)row * 128;
            float* orow = out + (size_t)row * 128;
#pragma unroll
            for (int nt = 0; nt < 8; ++nt) {
                int n = nt * 16 + mlo;
                yrow[n] = accn[nt][r];
                orow[n] = accs[nt][r] + bias[n];
            }
        }
    }
}

// ---------------------------------------------------------------------------
// out[dst, :] += w * y[src, :]; one wave32 per edge, float4 per lane,
// fp32 global atomics (complete in L2: y and out both fit in 192 MB).
// ---------------------------------------------------------------------------
__global__ __launch_bounds__(256) void spmm_scatter(
        const int* __restrict__ esrc, const int* __restrict__ edst,
        const float* __restrict__ ew, const float* __restrict__ y,
        float* __restrict__ out, int n_edges) {
    long gid = (long)blockIdx.x * blockDim.x + threadIdx.x;
    int e    = (int)(gid >> 5);
    int lane = (int)(gid & 31);
    if (e >= n_edges) return;
    int   s = esrc[e];
    int   d = edst[e];
    float w = ew[e];
    const f32x4* yv = (const f32x4*)(y + (size_t)s * 128);
    f32x4 v = yv[lane];
    float* o = out + (size_t)d * 128 + lane * 4;
    atomicAdd(o + 0, w * v.x);
    atomicAdd(o + 1, w * v.y);
    atomicAdd(o + 2, w * v.z);
    atomicAdd(o + 3, w * v.w);
}

// ---------------------------------------------------------------------------
extern "C" void kernel_launch(void* const* d_in, const int* in_sizes, int n_in,
                              void* d_out, int out_size, void* d_ws, size_t ws_size,
                              hipStream_t stream) {
    const float* x    = (const float*)d_in[0];
    const int*   esrc = (const int*)d_in[1];
    const int*   edst = (const int*)d_in[2];
    const float* ew   = (const float*)d_in[3];
    const float* Wn   = (const float*)d_in[4];
    const float* Ws   = (const float*)d_in[5];
    const float* bias = (const float*)d_in[6];

    const int n_nodes = in_sizes[0] / 128;
    const int n_edges = in_sizes[1];

    // Workspace: [4 contiguous bf16 W planes : 128 KB][y : n*128 f32]
    unsigned short* wplanes = (unsigned short*)d_ws;
    float*          y       = (float*)(wplanes + 4 * 128 * 128);

    prep_wT<<<64, 256, 0, stream>>>(Wn, Ws, wplanes);

    const int gblocks = (n_nodes + 127) / 128;
    gemm_dual_bf16x3<<<gblocks, 256, LDS_TOTAL_BYTES, stream>>>(
        x, wplanes, bias, y, (float*)d_out, n_nodes);

    const long total   = (long)n_edges * 32;
    const int  sblocks = (int)((total + 255) / 256);
    spmm_scatter<<<sblocks, 256, 0, stream>>>(esrc, edst, ew, y,
                                              (float*)d_out, n_edges);
}
